// BatchedAttn_Net_Gated_62637803045031
// MI455X (gfx1250) — matrix operand order repaired and checked by reading, not compiled, not run on hardware
//
#include <hip/hip_runtime.h>
#include <hip/hip_bf16.h>

typedef __attribute__((ext_vector_type(16))) _Float16 v16h;
typedef __attribute__((ext_vector_type(8)))  _Float16 h8;
typedef __attribute__((ext_vector_type(4)))  _Float16 h4;
typedef __attribute__((ext_vector_type(8)))  float    v8f;
typedef __attribute__((ext_vector_type(4)))  float    f4;

#define BATCH 32
#define NN    4096
#define LL    512
#define DD    256
#define BN    (BATCH*NN)     // 131072 rows
#define MT    128            // rows per block (halves L2 weight re-reads vs 64)
#define KT    32             // K tile
#define LDK   40             // padded LDS row stride (halfs): 80B -> conflict-free b128 frag loads
#define NTHR  512            // 16 waves: 4 (wm) x 4 (wn)

// ---------------------------------------------------------------------------
// Kernel 1: fused  a=tanh(xWa+ba), g=sigm(xWb+bb), s=(a*g)·Wc  + x passthrough
// ---------------------------------------------------------------------------
__global__ __launch_bounds__(NTHR) void k_gemm_score(
    const float* __restrict__ x,
    const float* __restrict__ Wa, const float* __restrict__ ba,
    const float* __restrict__ Wb, const float* __restrict__ bb,
    const float* __restrict__ Wc, const float* __restrict__ bc,
    float* __restrict__ scoreOut, float* __restrict__ xCopy)
{
    __shared__ __align__(16) _Float16 xs [MT*LDK];
    __shared__ __align__(16) _Float16 was[DD*LDK];
    __shared__ __align__(16) _Float16 wbs[DD*LDK];
    __shared__ float ssum[MT];

    const int tid  = threadIdx.x;
    const int lane = tid & 31;
    const int w    = tid >> 5;        // 0..15
    const int wm   = w >> 2;          // 0..3 : 32-row band
    const int wn   = w & 3;           // 0..3 : 64-col band
    const int m0   = blockIdx.x * MT;

    v8f acc_a[2][4];
    v8f acc_g[2][4];
    #pragma unroll
    for (int i = 0; i < 2; ++i)
        #pragma unroll
        for (int j = 0; j < 4; ++j) { acc_a[i][j] = v8f{}; acc_g[i][j] = v8f{}; }

    for (int kt = 0; kt < LL / KT; ++kt) {
        const int k0 = kt * KT;

        // ---- stage x tile (128x32): fp32 -> LDS f16, + verbatim copy to out
        #pragma unroll
        for (int it = 0; it < 2; ++it) {
            int idx = it * NTHR + tid;             // 1024 float4s = 128 rows x 8
            int m   = idx >> 3;
            int k   = (idx & 7) << 2;
            size_t g = (size_t)(m0 + m) * LL + k0 + k;
            f4 v = *(const f4*)(x + g);
            *(f4*)(xCopy + g) = v;                 // fused pass-through copy
            h4 hv; hv.x = (_Float16)v.x; hv.y = (_Float16)v.y;
                   hv.z = (_Float16)v.z; hv.w = (_Float16)v.w;
            *(h4*)&xs[m * LDK + k] = hv;
        }
        // ---- stage Wa/Wb tiles (32x256) -> transposed LDS [d][k] f16.
        //      Each task gathers a k-column (4 coalesced b32 loads, W is
        //      L2-resident) and stores one contiguous ds_store_b64.
        #pragma unroll
        for (int it = 0; it < 4; ++it) {
            int idx = it * NTHR + tid;             // 2048 tasks = 256 d x 8 k-blocks
            int d   = idx & (DD - 1);
            int k   = (idx >> 8) << 2;
            const float* pa = Wa + (size_t)(k0 + k) * DD + d;
            const float* pb = Wb + (size_t)(k0 + k) * DD + d;
            h4 ha, hb;
            #pragma unroll
            for (int i = 0; i < 4; ++i) {
                ha[i] = (_Float16)pa[(size_t)i * DD];
                hb[i] = (_Float16)pb[(size_t)i * DD];
            }
            *(h4*)&was[d * LDK + k] = ha;
            *(h4*)&wbs[d * LDK + k] = hb;
        }
        __syncthreads();

        // ---- A fragments (ISA 16-bit A 16x32 layout: elems 0..7 -> K=half*8+i,
        //      elems 8..15 -> K=16+half*8+i, row = lane%16)
        union U { v16h v; h8 h[2]; };
        U af[2];
        const int khA = (lane >> 4) << 3;
        #pragma unroll
        for (int ms = 0; ms < 2; ++ms) {
            int m = wm * 32 + ms * 16 + (lane & 15);
            af[ms].h[0] = *(const h8*)&xs[m * LDK + khA];
            af[ms].h[1] = *(const h8*)&xs[m * LDK + khA + 16];
        }
        // ---- B fragments (col = lane%16, elems 0..7 -> K=half*8+i, elems 8..15 -> K=16+half*8+i) + WMMA
        const int khB = (lane >> 4) << 3;
        #pragma unroll
        for (int ns = 0; ns < 4; ++ns) {
            int d = wn * 64 + ns * 16 + (lane & 15);
            U bfa, bfb;
            bfa.h[0] = *(const h8*)&was[d * LDK + khB];
            bfa.h[1] = *(const h8*)&was[d * LDK + khB + 16];
            bfb.h[0] = *(const h8*)&wbs[d * LDK + khB];
            bfb.h[1] = *(const h8*)&wbs[d * LDK + khB + 16];
            #pragma unroll
            for (int ms = 0; ms < 2; ++ms) {
                acc_a[ms][ns] = __builtin_amdgcn_wmma_f32_16x16x32_f16(
                    false, af[ms].v, false, bfa.v, (short)0, acc_a[ms][ns], false, false);
                acc_g[ms][ns] = __builtin_amdgcn_wmma_f32_16x16x32_f16(
                    false, af[ms].v, false, bfb.v, (short)0, acc_g[ms][ns], false, false);
            }
        }
        __syncthreads();
    }

    // ---- epilogue: s[m] = sum_d tanh(a+ba)*sigm(g+bb)*Wc[d]  (branchless)
    if (tid < MT) ssum[tid] = 0.0f;
    __syncthreads();
    const int half8 = (lane >> 4) << 3;   // C/D layout: M = r + (lane>=16 ? 8 : 0)
    #pragma unroll
    for (int ms = 0; ms < 2; ++ms) {
        float part[8];
        #pragma unroll
        for (int r = 0; r < 8; ++r) part[r] = 0.0f;
        #pragma unroll
        for (int ns = 0; ns < 4; ++ns) {
            int d = wn * 64 + ns * 16 + (lane & 15);
            float bav = ba[d], bbv = bb[d], wcv = Wc[d];
            #pragma unroll
            for (int r = 0; r < 8; ++r) {
                float za = acc_a[ms][ns][r] + bav;
                float zg = acc_g[ms][ns][r] + bbv;
                // tanh(z) = 1 - 2/(exp(2z)+1) ; sigmoid(z) = 1/(1+exp(-z))
                float aa = 1.0f - 2.0f * __builtin_amdgcn_rcpf(__expf(2.0f * za) + 1.0f);
                float gg = __builtin_amdgcn_rcpf(1.0f + __expf(-zg));
                part[r] += aa * gg * wcv;
            }
        }
        #pragma unroll
        for (int r = 0; r < 8; ++r) {
            int mloc = wm * 32 + ms * 16 + r + half8;
            atomicAdd(&ssum[mloc], part[r]);          // ds_add_f32
        }
    }
    __syncthreads();
    if (tid < MT) scoreOut[m0 + tid] = ssum[tid] + bc[0];
}

// ---------------------------------------------------------------------------
// Kernel 2: in-place softmax over N=4096 per batch
// ---------------------------------------------------------------------------
__global__ __launch_bounds__(256) void k_softmax(float* __restrict__ A)
{
    __shared__ float red[256];
    const int tid = threadIdx.x;
    float* row = A + (size_t)blockIdx.x * NN;

    float vals[16];
    float mx = -3.402823466e38f;
    #pragma unroll
    for (int i = 0; i < 16; ++i) { vals[i] = row[tid + i * 256]; mx = fmaxf(mx, vals[i]); }
    red[tid] = mx; __syncthreads();
    for (int s = 128; s > 0; s >>= 1) { if (tid < s) red[tid] = fmaxf(red[tid], red[tid + s]); __syncthreads(); }
    mx = red[0]; __syncthreads();

    float sum = 0.0f;
    #pragma unroll
    for (int i = 0; i < 16; ++i) { vals[i] = __expf(vals[i] - mx); sum += vals[i]; }
    red[tid] = sum; __syncthreads();
    for (int s = 128; s > 0; s >>= 1) { if (tid < s) red[tid] += red[tid + s]; __syncthreads(); }
    float inv = __builtin_amdgcn_rcpf(red[0]);
    #pragma unroll
    for (int i = 0; i < 16; ++i) row[tid + i * 256] = vals[i] * inv;
}

// ---------------------------------------------------------------------------
// Kernels 3/4: pooled[b,l] = sum_n A[b,n] * x[b,n,l]
// ---------------------------------------------------------------------------
__global__ void k_zero(float* __restrict__ p, int n)
{
    int i = blockIdx.x * blockDim.x + threadIdx.x;
    if (i < n) p[i] = 0.0f;
}

__global__ __launch_bounds__(128) void k_pool(
    const float* __restrict__ x, const float* __restrict__ A,
    float* __restrict__ pooled)
{
    const int b  = blockIdx.x;           // 0..31
    const int nc = blockIdx.y;           // 0..15 (256 n each)
    const int l  = threadIdx.x << 2;     // float4 per thread covers L=512
    const float* xb = x + ((size_t)b * NN + (size_t)nc * 256) * LL;
    const float* Ab = A + (size_t)b * NN + nc * 256;
    f4 acc = f4{};
    for (int i = 0; i < 256; ++i) {
        float wgt = Ab[i];
        f4 v = *(const f4*)(xb + (size_t)i * LL + l);
        acc += v * wgt;
    }
    float* o = pooled + b * LL + l;
    atomicAdd(o + 0, acc.x);
    atomicAdd(o + 1, acc.y);
    atomicAdd(o + 2, acc.z);
    atomicAdd(o + 3, acc.w);
}

// ---------------------------------------------------------------------------
extern "C" void kernel_launch(void* const* d_in, const int* in_sizes, int n_in,
                              void* d_out, int out_size, void* d_ws, size_t ws_size,
                              hipStream_t stream)
{
    const float* x  = (const float*)d_in[0];
    const float* Wa = (const float*)d_in[1];
    const float* ba = (const float*)d_in[2];
    const float* Wb = (const float*)d_in[3];
    const float* bb = (const float*)d_in[4];
    const float* Wc = (const float*)d_in[5];
    const float* bc = (const float*)d_in[6];

    float* out    = (float*)d_out;
    float* A      = out;                       // [B,N]   131072 floats
    float* xCopy  = A + (size_t)BN;            // [B,N,L] 67108864 floats
    float* pooled = xCopy + (size_t)BN * LL;   // [B,L]   16384 floats

    k_gemm_score<<<BN / MT, NTHR, 0, stream>>>(x, Wa, ba, Wb, bb, Wc, bc, A, xCopy);
    k_softmax<<<BATCH, 256, 0, stream>>>(A);
    k_zero<<<(BATCH * LL + 255) / 256, 256, 0, stream>>>(pooled, BATCH * LL);
    k_pool<<<dim3(BATCH, 16), 128, 0, stream>>>(x, A, pooled);
}
